// LocalGlobalBlock_12773232739142
// MI455X (gfx1250) — matrix operand order repaired and checked by reading, not compiled, not run on hardware
//
#include <hip/hip_runtime.h>
#include <cstdint>
#include <cstddef>

typedef _Float16 f16;
typedef f16   v16h __attribute__((ext_vector_type(16)));
typedef float v8f  __attribute__((ext_vector_type(8)));

// ---------------------------------------------------------------- WMMA utils
__device__ __forceinline__ v8f wmma16(v16h a, v16h b, v8f c) {
  // D = A(16x32 f16) * B(32x16 f16) + C(16x16 f32)
  return __builtin_amdgcn_wmma_f32_16x16x32_f16(false, a, false, b, (short)0, c,
                                                false, false);
}
__device__ __forceinline__ v8f vzero8() {
  v8f z = {0.f, 0.f, 0.f, 0.f, 0.f, 0.f, 0.f, 0.f};
  return z;
}
// A fragment (16x32, row-major source, ld in halves).
// lanes 0-15: row m, K slots {k0..k0+7, k0+16..k0+23}; lanes 16-31: +8 on K.
__device__ __forceinline__ v16h ldA(const f16* base, int ld, int row0, int k0) {
  int lane = threadIdx.x & 31;
  const f16* p = base + (size_t)(row0 + (lane & 15)) * ld + k0 + ((lane >> 4) << 3);
  union { v16h v; uint4 q[2]; } u;
  u.q[0] = *(const uint4*)p;
  u.q[1] = *(const uint4*)(p + 16);
  return u.v;
}
// B fragment from Bt[N][K] (row-major, ld in halves).
// lane n = col0 + (lane&15); lanes 0-15 hold K {k0..k0+7, k0+16..k0+23}; lanes 16-31: +8 on K.
__device__ __forceinline__ v16h ldB(const f16* bt, int ld, int col0, int k0) {
  int lane = threadIdx.x & 31;
  const f16* p = bt + (size_t)(col0 + (lane & 15)) * ld + k0 + ((lane >> 4) << 3);
  union { v16h v; uint4 q[2]; } u;
  u.q[0] = *(const uint4*)p;
  u.q[1] = *(const uint4*)(p + 16);
  return u.v;
}
__device__ __forceinline__ float rmax16(float v) {
#pragma unroll
  for (int m = 1; m < 16; m <<= 1) v = fmaxf(v, __shfl_xor(v, m, 32));
  return v;
}
__device__ __forceinline__ float rsum16(float v) {
#pragma unroll
  for (int m = 1; m < 16; m <<= 1) v += __shfl_xor(v, m, 32);
  return v;
}

// ------------------------------------------------------- weight f32 -> f16^T
__global__ __launch_bounds__(256) void k_wtr(const float* __restrict__ W,
                                             f16* __restrict__ Wt, int K, int N) {
  size_t idx = (size_t)blockIdx.x * 256 + threadIdx.x;  // over N*K
  int n = (int)(idx / K);
  int k = (int)(idx % K);
  Wt[idx] = (f16)W[(size_t)k * N + n];
}

// ------------------------------------------------- build XC = [msg ; region x]
// XC rows: r*99 + 0 = messenger, r*99 + 1 + s = region token s. r = b*32 + reg.
__global__ __launch_bounds__(256) void k_build_xc(const float* __restrict__ x,
                                                  const float* __restrict__ msg,
                                                  float* __restrict__ XC) {
  size_t idx = (size_t)blockIdx.x * 256 + threadIdx.x;  // float4 units, 50688*128
  size_t rowq = idx >> 7;
  int q = (int)(idx & 127);
  int r = (int)(rowq / 99), s99 = (int)(rowq % 99);
  const float* src;
  if (s99 == 0) {
    src = msg + (size_t)r * 512;
  } else {
    int s = s99 - 1;
    int bI = r >> 5, reg = r & 31;
    int nt_i = reg >> 2, nh_i = (reg >> 1) & 1, nw_i = reg & 1;
    int rt_i = s / 49, rem = s % 49, rh_i = rem / 7, rw_i = rem % 7;
    int tt = nt_i * 2 + rt_i, hh = nh_i * 7 + rh_i, ww = nw_i * 7 + rw_i;
    int n = (tt * 14 + hh) * 14 + ww;
    src = x + ((size_t)bI * 3136 + n) * 512;
  }
  ((float4*)XC)[rowq * 128 + q] = ((const float4*)src)[q];
}

// ------------------------------------------------- split XC -> local | msg
__global__ __launch_bounds__(256) void k_split_xc(const float* __restrict__ XC,
                                                  float* __restrict__ outL,
                                                  float* __restrict__ msg) {
  size_t idx = (size_t)blockIdx.x * 256 + threadIdx.x;  // float4 units
  size_t rowq = idx >> 7;
  int q = (int)(idx & 127);
  int r = (int)(rowq / 99), s99 = (int)(rowq % 99);
  float4 v = ((const float4*)XC)[idx];
  if (s99 == 0)
    ((float4*)msg)[(size_t)r * 128 + q] = v;
  else
    ((float4*)outL)[((size_t)r * 98 + (s99 - 1)) * 128 + q] = v;
}

// ----------------------------------------------------------------- layernorm
__global__ __launch_bounds__(256) void k_ln(const float* __restrict__ X,
                                            const float* __restrict__ g,
                                            const float* __restrict__ b,
                                            f16* __restrict__ Y, int rows) {
  int lane = threadIdx.x & 31, wave = threadIdx.x >> 5;
  int row = blockIdx.x * 8 + wave;
  if (row >= rows) return;
  const float* x = X + (size_t)row * 512;
  float v[16], s = 0.f, ss = 0.f;
#pragma unroll
  for (int i = 0; i < 16; ++i) {
    float t = x[i * 32 + lane];
    v[i] = t; s += t; ss += t * t;
  }
#pragma unroll
  for (int m = 1; m < 32; m <<= 1) {
    s  += __shfl_xor(s, m, 32);
    ss += __shfl_xor(ss, m, 32);
  }
  float mean = s * (1.f / 512.f);
  float var  = ss * (1.f / 512.f) - mean * mean;
  float rstd = rsqrtf(var + 1e-5f);
  f16* y = Y + (size_t)row * 512;
#pragma unroll
  for (int i = 0; i < 16; ++i) {
    int c = i * 32 + lane;
    y[c] = (f16)((v[i] - mean) * rstd * g[c] + b[c]);
  }
}

// ------------------------------------------------------------- generic GEMM
// out = A(MxK f16) * Bt^T + bias + resid ; wave computes 16x64 strip.
// Branchless software pipeline: fragments for k+32 are prefetched
// unconditionally (the final prefetch overreads one k-step, which stays inside
// the workspace / returns zero from LDS and is never consumed).
__global__ __launch_bounds__(256) void k_gemm(const f16* __restrict__ A, int lda,
                                              const f16* __restrict__ Bt,
                                              const float* __restrict__ bias,
                                              const float* __restrict__ resid,
                                              float* __restrict__ outF,
                                              f16* __restrict__ outH,
                                              int M, int N, int K) {
  int lane = threadIdx.x & 31, wave = threadIdx.x >> 5;
  int strips = N >> 6;
  int gid = blockIdx.x * 8 + wave;
  if (gid >= (M >> 4) * strips) return;
  int m0 = (gid / strips) << 4;
  int n0 = (gid % strips) << 6;
  v8f acc0 = vzero8(), acc1 = vzero8(), acc2 = vzero8(), acc3 = vzero8();

  v16h a_cur = ldA(A, lda, m0, 0);
  v16h b0 = ldB(Bt, K, n0, 0);
  v16h b1 = ldB(Bt, K, n0 + 16, 0);
  v16h b2 = ldB(Bt, K, n0 + 32, 0);
  v16h b3 = ldB(Bt, K, n0 + 48, 0);

  for (int k0 = 0; k0 < K; k0 += 32) {
    int kn = k0 + 32;  // last iteration overreads one step: harmless
    v16h a_n  = ldA(A, lda, m0, kn);
    v16h b0n = ldB(Bt, K, n0, kn);
    v16h b1n = ldB(Bt, K, n0 + 16, kn);
    v16h b2n = ldB(Bt, K, n0 + 32, kn);
    v16h b3n = ldB(Bt, K, n0 + 48, kn);
    acc0 = wmma16(a_cur, b0, acc0);
    acc1 = wmma16(a_cur, b1, acc1);
    acc2 = wmma16(a_cur, b2, acc2);
    acc3 = wmma16(a_cur, b3, acc3);
    a_cur = a_n; b0 = b0n; b1 = b1n; b2 = b2n; b3 = b3n;
  }

  int ln = lane & 15, hi = lane >> 4;
  v8f accs[4] = {acc0, acc1, acc2, acc3};
#pragma unroll
  for (int j = 0; j < 4; ++j) {
    int col = n0 + j * 16 + ln;
    float bc = bias ? bias[col] : 0.f;
#pragma unroll
    for (int g = 0; g < 8; ++g) {
      int row = m0 + g + hi * 8;
      float v = accs[j][g] + bc;
      if (resid) v += resid[(size_t)row * N + col];
      if (outF) outF[(size_t)row * N + col] = v;
      if (outH) outH[(size_t)row * N + col] = (f16)v;
    }
  }
}

// ------------------------------------------- stage-1 local attention (L=99)
// block = (region, head); seq padded 99->112, K-dim of PV padded to 128.
__global__ __launch_bounds__(256) void k_attn_local(const f16* __restrict__ QKV,
                                                    f16* __restrict__ O) {
  __shared__ __align__(16) f16 smA[14336];  // Q[112*64] | K[112*64] ; then P[112*128]
  __shared__ __align__(16) f16 smV[8192];   // Vt[64][128]
  f16* sQ = smA;
  f16* sK = smA + 7168;
  f16* sP = smA;
  f16* sVt = smV;
  const int reg = blockIdx.x >> 3;
  const int hD  = blockIdx.x & 7;
  const int lane = threadIdx.x & 31, wave = threadIdx.x >> 5;
  const int ln = lane & 15, hi = lane >> 4;
  const f16* qkv = QKV + (size_t)reg * 99 * 1536 + hD * 64;

  for (int c = threadIdx.x; c < 112 * 8; c += 256) {
    int seq = c >> 3, dc = (c & 7) << 3;
    uint4 vq = {0, 0, 0, 0}, vk = vq;
    if (seq < 99) {
      const f16* row = qkv + (size_t)seq * 1536 + dc;
      vq = *(const uint4*)row;
      vk = *(const uint4*)(row + 512);
    }
    *(uint4*)&sQ[seq * 64 + dc] = vq;
    *(uint4*)&sK[seq * 64 + dc] = vk;
  }
  for (int i = threadIdx.x; i < 112 * 64; i += 256) {
    int seq = i >> 6, d = i & 63;
    f16 v = (f16)0.f;
    if (seq < 99) v = qkv[(size_t)seq * 1536 + 1024 + d];
    sVt[d * 128 + seq] = v;
  }
  for (int i = threadIdx.x; i < 64 * 16; i += 256)
    sVt[(i >> 4) * 128 + 112 + (i & 15)] = (f16)0.f;
  __syncthreads();

  v8f ps[7];
  const int m0 = wave * 16;
  if (wave < 7) {
#pragma unroll
    for (int ct = 0; ct < 7; ++ct) {
      v8f acc = vzero8();
#pragma unroll
      for (int k0 = 0; k0 < 64; k0 += 32)
        acc = wmma16(ldA(sQ, 64, m0, k0), ldB(sK, 64, ct * 16, k0), acc);
      ps[ct] = acc;
    }
#pragma unroll
    for (int ct = 0; ct < 7; ++ct) {
      bool valid = (ct * 16 + ln) < 99;
#pragma unroll
      for (int g = 0; g < 8; ++g) {
        float v = ps[ct][g] * 0.125f;  // hd^-0.5
        ps[ct][g] = valid ? v : -__builtin_inff();
      }
    }
#pragma unroll
    for (int g = 0; g < 8; ++g) {
      float mx = -__builtin_inff();
#pragma unroll
      for (int ct = 0; ct < 7; ++ct) mx = fmaxf(mx, ps[ct][g]);
      mx = rmax16(mx);
      float sum = 0.f;
#pragma unroll
      for (int ct = 0; ct < 7; ++ct) {
        float e = __expf(ps[ct][g] - mx);
        ps[ct][g] = e; sum += e;
      }
      float inv = 1.f / rsum16(sum);
#pragma unroll
      for (int ct = 0; ct < 7; ++ct) ps[ct][g] *= inv;
    }
  }
  __syncthreads();  // everyone done reading sQ/sK before P overwrites them
  if (wave < 7) {
#pragma unroll
    for (int ct = 0; ct < 7; ++ct)
#pragma unroll
      for (int g = 0; g < 8; ++g)
        sP[(m0 + g + hi * 8) * 128 + ct * 16 + ln] = (f16)ps[ct][g];
#pragma unroll
    for (int g = 0; g < 8; ++g)
      sP[(m0 + g + hi * 8) * 128 + 112 + ln] = (f16)0.f;
  }
  __syncthreads();
  if (wave < 7) {
    v8f o[4] = {vzero8(), vzero8(), vzero8(), vzero8()};
#pragma unroll
    for (int k0 = 0; k0 < 128; k0 += 32) {
      v16h a = ldA(sP, 128, m0, k0);
#pragma unroll
      for (int dt = 0; dt < 4; ++dt)
        o[dt] = wmma16(a, ldB(sVt, 128, dt * 16, k0), o[dt]);
    }
#pragma unroll
    for (int dt = 0; dt < 4; ++dt)
#pragma unroll
      for (int g = 0; g < 8; ++g) {
        int seq = m0 + g + hi * 8;
        if (seq < 99)
          O[((size_t)reg * 99 + seq) * 512 + hD * 64 + dt * 16 + ln] = (f16)o[dt][g];
      }
  }
}

// ------------------------------------- stage-2 messenger attention (32x32)
__global__ __launch_bounds__(256) void k_attn_msg(const f16* __restrict__ QKV,
                                                  f16* __restrict__ O) {
  __shared__ __align__(16) f16 sPa[8][1024];   // per-wave P 32x32
  __shared__ __align__(16) f16 sVa[8][2048];   // per-wave Vt 64x32
  const int lane = threadIdx.x & 31, wave = threadIdx.x >> 5;
  const int ln = lane & 15, hi = lane >> 4;
  int wid = blockIdx.x * 8 + wave;   // 128 = 16 batches x 8 heads
  int bI = wid >> 3, hD = wid & 7;
  f16* sP = sPa[wave];
  f16* sVt = sVa[wave];
  const f16* base = QKV + (size_t)bI * 32 * 1536;
  const f16* Qb = base + hD * 64;
  const f16* Kb = base + 512 + hD * 64;
  const f16* Vb = base + 1024 + hD * 64;
  for (int i = lane; i < 2048; i += 32) {
    int seq = i & 31, d = i >> 5;
    sVt[d * 32 + seq] = Vb[(size_t)seq * 1536 + d];
  }
#pragma unroll
  for (int rt = 0; rt < 2; ++rt) {
    v8f s2[2] = {vzero8(), vzero8()};
#pragma unroll
    for (int k0 = 0; k0 < 64; k0 += 32) {
      v16h a = ldA(Qb, 1536, rt * 16, k0);
#pragma unroll
      for (int ct = 0; ct < 2; ++ct)
        s2[ct] = wmma16(a, ldB(Kb, 1536, ct * 16, k0), s2[ct]);
    }
#pragma unroll
    for (int g = 0; g < 8; ++g) {
      float a0 = s2[0][g] * 0.125f, a1 = s2[1][g] * 0.125f;
      float mx = rmax16(fmaxf(a0, a1));
      float e0 = __expf(a0 - mx), e1 = __expf(a1 - mx);
      float inv = 1.f / rsum16(e0 + e1);
      s2[0][g] = e0 * inv; s2[1][g] = e1 * inv;
    }
#pragma unroll
    for (int ct = 0; ct < 2; ++ct)
#pragma unroll
      for (int g = 0; g < 8; ++g)
        sP[(rt * 16 + g + hi * 8) * 32 + ct * 16 + ln] = (f16)s2[ct][g];
    v8f o[4] = {vzero8(), vzero8(), vzero8(), vzero8()};
    v16h a2 = ldA(sP, 32, rt * 16, 0);
#pragma unroll
    for (int dt = 0; dt < 4; ++dt)
      o[dt] = wmma16(a2, ldB(sVt, 32, dt * 16, 0), o[dt]);
#pragma unroll
    for (int dt = 0; dt < 4; ++dt)
#pragma unroll
      for (int g = 0; g < 8; ++g)
        O[((size_t)bI * 32 + rt * 16 + g + hi * 8) * 512 + hD * 64 + dt * 16 + ln] =
            (f16)o[dt][g];
  }
}

// ------------------------ stage-3 cross attention (16 queries x 32 kv / wave)
__global__ __launch_bounds__(256) void k_attn_cross(const f16* __restrict__ Q,
                                                    const f16* __restrict__ KV,
                                                    f16* __restrict__ O) {
  __shared__ __align__(16) f16 sPa[8][512];    // per-wave P 16x32
  __shared__ __align__(16) f16 sVa[8][2048];   // per-wave Vt 64x32
  const int lane = threadIdx.x & 31, wave = threadIdx.x >> 5;
  const int ln = lane & 15, hi = lane >> 4;
  int wid = blockIdx.x * 8 + wave;   // 25088 = 196 qtiles x 16 b x 8 h
  int qt = wid % 196;
  int bh = wid / 196;
  int bI = bh >> 3, hD = bh & 7;
  f16* sP = sPa[wave];
  f16* sVt = sVa[wave];
  const f16* Qb = Q + ((size_t)bI * 3136 + qt * 16) * 512 + hD * 64;
  const f16* Kb = KV + (size_t)bI * 32 * 1024 + hD * 64;
  const f16* Vb = Kb + 512;
  for (int i = lane; i < 2048; i += 32) {
    int seq = i & 31, d = i >> 5;
    sVt[d * 32 + seq] = Vb[(size_t)seq * 1024 + d];
  }
  v8f s2[2] = {vzero8(), vzero8()};
#pragma unroll
  for (int k0 = 0; k0 < 64; k0 += 32) {
    v16h a = ldA(Qb, 512, 0, k0);
#pragma unroll
    for (int ct = 0; ct < 2; ++ct)
      s2[ct] = wmma16(a, ldB(Kb, 1024, ct * 16, k0), s2[ct]);
  }
#pragma unroll
  for (int g = 0; g < 8; ++g) {
    float a0 = s2[0][g] * 0.125f, a1 = s2[1][g] * 0.125f;
    float mx = rmax16(fmaxf(a0, a1));
    float e0 = __expf(a0 - mx), e1 = __expf(a1 - mx);
    float inv = 1.f / rsum16(e0 + e1);
    s2[0][g] = e0 * inv; s2[1][g] = e1 * inv;
  }
#pragma unroll
  for (int ct = 0; ct < 2; ++ct)
#pragma unroll
    for (int g = 0; g < 8; ++g)
      sP[(g + hi * 8) * 32 + ct * 16 + ln] = (f16)s2[ct][g];
  v8f o[4] = {vzero8(), vzero8(), vzero8(), vzero8()};
  v16h a2 = ldA(sP, 32, 0, 0);
#pragma unroll
  for (int dt = 0; dt < 4; ++dt)
    o[dt] = wmma16(a2, ldB(sVt, 32, dt * 16, 0), o[dt]);
#pragma unroll
  for (int dt = 0; dt < 4; ++dt)
#pragma unroll
    for (int g = 0; g < 8; ++g)
      O[((size_t)bI * 3136 + qt * 16 + g + hi * 8) * 512 + hD * 64 + dt * 16 + ln] =
          (f16)o[dt][g];
}

// --------------------------------------------------------------- fused MLP
// out = gelu_exact(Xln @ W1 + b1) @ W2 + b2 + resid ; block = 16 rows.
// hid never leaves LDS; both inner GEMMs use the branchless prefetch pipeline
// (one-step overread past K: weights stay inside workspace, LDS returns zero).
__global__ __launch_bounds__(256) void k_mlp(const f16* __restrict__ Xln,
                                             const f16* __restrict__ W1t,  // [2048][512]
                                             const float* __restrict__ b1,
                                             const f16* __restrict__ W2t,  // [512][2048]
                                             const float* __restrict__ b2,
                                             const float* __restrict__ resid,
                                             float* __restrict__ out) {
  __shared__ __align__(16) f16 sX[16 * 512];
  __shared__ __align__(16) f16 sH[16 * 512];
  const int lane = threadIdx.x & 31, wave = threadIdx.x >> 5;
  const int ln = lane & 15, hi = lane >> 4;
  const int m0 = blockIdx.x * 16;
  for (int c = threadIdx.x; c < 1024; c += 256) {
    int r = c >> 6, dc = (c & 63) << 3;
    *(uint4*)&sX[r * 512 + dc] = *(const uint4*)&Xln[(size_t)(m0 + r) * 512 + dc];
  }
  __syncthreads();
  v8f acc0 = vzero8(), acc1 = vzero8(), acc2 = vzero8(), acc3 = vzero8();
  for (int hc = 0; hc < 4; ++hc) {
    const f16* w1base = W1t + (size_t)(hc * 512 + wave * 64) * 512;
    v8f ha0 = vzero8(), ha1 = vzero8(), ha2 = vzero8(), ha3 = vzero8();
    {
      v16h a_cur = ldA(sX, 512, 0, 0);
      v16h b0 = ldB(w1base, 512, 0, 0);
      v16h b1f = ldB(w1base, 512, 16, 0);
      v16h b2f = ldB(w1base, 512, 32, 0);
      v16h b3f = ldB(w1base, 512, 48, 0);
      for (int k0 = 0; k0 < 512; k0 += 32) {
        int kn = k0 + 32;  // final prefetch overreads one step: harmless
        v16h a_n  = ldA(sX, 512, 0, kn);
        v16h b0n = ldB(w1base, 512, 0, kn);
        v16h b1n = ldB(w1base, 512, 16, kn);
        v16h b2n = ldB(w1base, 512, 32, kn);
        v16h b3n = ldB(w1base, 512, 48, kn);
        ha0 = wmma16(a_cur, b0, ha0);
        ha1 = wmma16(a_cur, b1f, ha1);
        ha2 = wmma16(a_cur, b2f, ha2);
        ha3 = wmma16(a_cur, b3f, ha3);
        a_cur = a_n; b0 = b0n; b1f = b1n; b2f = b2n; b3f = b3n;
      }
    }
    __syncthreads();  // previous chunk's phase-B readers done with sH
    v8f has[4] = {ha0, ha1, ha2, ha3};
#pragma unroll
    for (int j = 0; j < 4; ++j) {
      int colL = wave * 64 + j * 16 + ln;
      float bb = b1[hc * 512 + colL];
#pragma unroll
      for (int g = 0; g < 8; ++g) {
        float h = has[j][g] + bb;
        h = 0.5f * h * (1.f + erff(h * 0.70710678118f));  // exact gelu
        sH[(g + hi * 8) * 512 + colL] = (f16)h;
      }
    }
    __syncthreads();
    {
      const f16* w2base = W2t + (size_t)(wave * 64) * 2048 + hc * 512;
      v16h a_cur = ldA(sH, 512, 0, 0);
      v16h b0 = ldB(w2base, 2048, 0, 0);
      v16h b1f = ldB(w2base, 2048, 16, 0);
      v16h b2f = ldB(w2base, 2048, 32, 0);
      v16h b3f = ldB(w2base, 2048, 48, 0);
      for (int k0 = 0; k0 < 512; k0 += 32) {
        int kn = k0 + 32;  // final prefetch overreads one step: harmless
        v16h a_n  = ldA(sH, 512, 0, kn);
        v16h b0n = ldB(w2base, 2048, 0, kn);
        v16h b1n = ldB(w2base, 2048, 16, kn);
        v16h b2n = ldB(w2base, 2048, 32, kn);
        v16h b3n = ldB(w2base, 2048, 48, kn);
        acc0 = wmma16(a_cur, b0, acc0);
        acc1 = wmma16(a_cur, b1f, acc1);
        acc2 = wmma16(a_cur, b2f, acc2);
        acc3 = wmma16(a_cur, b3f, acc3);
        a_cur = a_n; b0 = b0n; b1f = b1n; b2f = b2n; b3f = b3n;
      }
    }
  }
  v8f accs[4] = {acc0, acc1, acc2, acc3};
#pragma unroll
  for (int j = 0; j < 4; ++j) {
    int col = wave * 64 + j * 16 + ln;
    float bb = b2[col];
#pragma unroll
    for (int g = 0; g < 8; ++g) {
      int row = m0 + g + hi * 8;
      out[(size_t)row * 512 + col] = accs[j][g] + bb + resid[(size_t)row * 512 + col];
    }
  }
}

// =========================================================== host launcher
extern "C" void kernel_launch(void* const* d_in, const int* in_sizes, int n_in,
                              void* d_out, int out_size, void* d_ws, size_t ws_size,
                              hipStream_t stream) {
  // dims
  const int NC1 = 50688;   // 512 regions * 99
  const int NL  = 50176;   // 512 regions * 98

  // weight-transpose table (offsets in halves inside WT region)
  const size_t H_WQKV1 = 0,       H_WPROJ1 = 786432,  H_WQKV2 = 1048576;
  const size_t H_WPROJ2 = 1835008, H_WQ3 = 2097152,   H_WKV3 = 2359296;
  const size_t H_WPROJ3 = 2883584, H_WFC1 = 3145728,  H_WFC2 = 4194304;
  const size_t H_WMFC1 = 5242880,  H_WMFC2 = 6291456;

  const size_t OFF_XC = 29360128, OFF_LN16 = 133169152, OFF_QKV16 = 185073664;
  const size_t OFF_ATTN16 = 340787200, OFF_MSG = 392691712;
  const size_t OFF_MSGLN = 393740288, OFF_MSGQKV = 394264576, OFF_MSGATT = 395837440;
  const size_t WS_NEEDED = 396361728 + 4096;  // + slack for pipeline overread
  if (ws_size < WS_NEEDED) return;

  const float* x    = (const float*)d_in[0];
  const float* mtok = (const float*)d_in[1];
  const float* g1a = (const float*)d_in[2],  *b1a = (const float*)d_in[3];
  const float* bproj1 = (const float*)d_in[6];
  const float* g2a = (const float*)d_in[7],  *b2a = (const float*)d_in[8];
  const float* bproj2 = (const float*)d_in[11];
  const float* g3a = (const float*)d_in[12], *b3a = (const float*)d_in[13];
  const float* g3c = (const float*)d_in[14], *b3c = (const float*)d_in[15];
  const float* bproj3 = (const float*)d_in[19];
  const float* gn2 = (const float*)d_in[20], *bn2 = (const float*)d_in[21];
  const float* bfc1 = (const float*)d_in[23], *bfc2 = (const float*)d_in[25];
  const float* gm2 = (const float*)d_in[26], *bm2 = (const float*)d_in[27];
  const float* bmfc1 = (const float*)d_in[29], *bmfc2 = (const float*)d_in[31];

  char* wsb = (char*)d_ws;
  f16*   WT     = (f16*)wsb;
  float* XC     = (float*)(wsb + OFF_XC);
  f16*   LN16   = (f16*)(wsb + OFF_LN16);
  f16*   QKV16  = (f16*)(wsb + OFF_QKV16);   // stage1 qkv; reused as Q16 in stage3
  f16*   ATTN16 = (f16*)(wsb + OFF_ATTN16);
  float* MSG    = (float*)(wsb + OFF_MSG);
  f16*   MSGLN  = (f16*)(wsb + OFF_MSGLN);
  f16*   MSGQKV = (f16*)(wsb + OFF_MSGQKV);  // reused as KV16 in stage3
  f16*   MSGATT = (f16*)(wsb + OFF_MSGATT);
  float* OUTL = (float*)d_out;
  float* OUTM = OUTL + (size_t)25690112;

  // 1) weight conversions: W[K][N] f32 -> Wt[N][K] f16
  struct WTR { int idx; size_t off; int K, N; };
  const WTR tw[11] = {
      {4, H_WQKV1, 512, 1536},  {5, H_WPROJ1, 512, 512},
      {9, H_WQKV2, 512, 1536},  {10, H_WPROJ2, 512, 512},
      {16, H_WQ3, 512, 512},    {17, H_WKV3, 512, 1024},
      {18, H_WPROJ3, 512, 512}, {22, H_WFC1, 512, 2048},
      {24, H_WFC2, 2048, 512},  {28, H_WMFC1, 512, 2048},
      {30, H_WMFC2, 2048, 512}};
  for (int i = 0; i < 11; ++i)
    k_wtr<<<(tw[i].K * tw[i].N) / 256, 256, 0, stream>>>(
        (const float*)d_in[tw[i].idx], WT + tw[i].off, tw[i].K, tw[i].N);

  // 2) gather x + messengers into region layout
  k_build_xc<<<25344, 256, 0, stream>>>(x, mtok, XC);

  // ---- stage 1: local self-attention
  k_ln<<<NC1 / 8, 256, 0, stream>>>(XC, g1a, b1a, LN16, NC1);
  k_gemm<<<(NC1 / 16) * (1536 / 64) / 8, 256, 0, stream>>>(
      LN16, 512, WT + H_WQKV1, nullptr, nullptr, nullptr, QKV16, NC1, 1536, 512);
  k_attn_local<<<512 * 8, 256, 0, stream>>>(QKV16, ATTN16);
  k_gemm<<<(NC1 / 16) * (512 / 64) / 8, 256, 0, stream>>>(
      ATTN16, 512, WT + H_WPROJ1, bproj1, XC, XC, nullptr, NC1, 512, 512);
  k_split_xc<<<25344, 256, 0, stream>>>(XC, OUTL, MSG);

  // ---- stage 2: messenger self-attention
  k_ln<<<64, 256, 0, stream>>>(MSG, g2a, b2a, MSGLN, 512);
  k_gemm<<<(512 / 16) * (1536 / 64) / 8, 256, 0, stream>>>(
      MSGLN, 512, WT + H_WQKV2, nullptr, nullptr, nullptr, MSGQKV, 512, 1536, 512);
  k_attn_msg<<<16, 256, 0, stream>>>(MSGQKV, MSGATT);
  k_gemm<<<(512 / 16) * (512 / 64) / 8, 256, 0, stream>>>(
      MSGATT, 512, WT + H_WPROJ2, bproj2, MSG, MSG, nullptr, 512, 512, 512);

  // ---- stage 3: cross attention (local q, messenger kv)
  k_ln<<<NL / 8, 256, 0, stream>>>(OUTL, g3a, b3a, LN16, NL);
  k_ln<<<64, 256, 0, stream>>>(MSG, g3c, b3c, MSGLN, 512);
  k_gemm<<<(NL / 16) * (512 / 64) / 8, 256, 0, stream>>>(
      LN16, 512, WT + H_WQ3, nullptr, nullptr, nullptr, QKV16, NL, 512, 512);
  k_gemm<<<(512 / 16) * (1024 / 64) / 8, 256, 0, stream>>>(
      MSGLN, 512, WT + H_WKV3, nullptr, nullptr, nullptr, MSGQKV, 512, 1024, 512);
  k_attn_cross<<<3136, 256, 0, stream>>>(QKV16, MSGQKV, ATTN16);
  k_gemm<<<(NL / 16) * (512 / 64) / 8, 256, 0, stream>>>(
      ATTN16, 512, WT + H_WPROJ3, bproj3, OUTL, OUTL, nullptr, NL, 512, 512);

  // ---- MLPs
  k_ln<<<NL / 8, 256, 0, stream>>>(OUTL, gn2, bn2, LN16, NL);
  k_mlp<<<NL / 16, 256, 0, stream>>>(LN16, WT + H_WFC1, bfc1, WT + H_WFC2, bfc2,
                                     OUTL, OUTL);
  k_ln<<<64, 256, 0, stream>>>(MSG, gm2, bm2, MSGLN, 512);
  k_mlp<<<32, 256, 0, stream>>>(MSGLN, WT + H_WMFC1, bmfc1, WT + H_WMFC2, bmfc2,
                                MSG, OUTM);
}
